// WinMSA_84353157693458
// MI455X (gfx1250) — matrix operand order repaired and checked by reading, not compiled, not run on hardware
//
#include <hip/hip_runtime.h>
#include <cstdint>

// ---------------------------------------------------------------------------
// Types for CDNA5 WMMA (wave32): v_wmma_f32_16x16x32_bf16
// ---------------------------------------------------------------------------
typedef __bf16 bf8v  __attribute__((ext_vector_type(8)));
typedef __bf16 bf16v __attribute__((ext_vector_type(16)));
typedef float  v8f   __attribute__((ext_vector_type(8)));
typedef int    v2i   __attribute__((ext_vector_type(2)));

// may_alias variants: used when reinterpreting f32/u16 LDS as bf16 vectors so
// the compiler cannot TBAA-reorder the softmax read-f32/write-u16 overlay.
typedef bf8v           bf8v_a  __attribute__((may_alias));
typedef bf16v          bf16v_a __attribute__((may_alias));
typedef unsigned short u16;
typedef unsigned short u16_a   __attribute__((may_alias));

#define WMMA_BF16(A, B, C) \
  __builtin_amdgcn_wmma_f32_16x16x32_bf16(false, (A), false, (B), (short)0, (C), false, false)

// gfx1250 async global->LDS copy (ASYNCcnt-tracked, no VGPR round trip).
#if defined(__gfx1250__) && __has_builtin(__builtin_amdgcn_global_load_async_to_lds_b64)
#define HAS_ASYNC_LDS 1
#else
#define HAS_ASYNC_LDS 0
#endif

#if HAS_ASYNC_LDS
typedef __attribute__((address_space(1))) v2i* glb_p;  // prints as "__device__ v2i*"
typedef __attribute__((address_space(3))) v2i* lds_p;
__device__ __forceinline__ void async_copy_b64(const void* g, const void* l) {
  // Generic LDS addresses are {SHARED_BASE, offset32}; truncation yields the
  // 32-bit LDS offset (ISA aperture rules). Integer round-trips avoid
  // addrspacecast semantics entirely.
  __builtin_amdgcn_global_load_async_to_lds_b64(
      (glb_p)(uintptr_t)g, (lds_p)(uintptr_t)(unsigned)(uintptr_t)l, 0, 0);
}
__device__ __forceinline__ void async_wait() {
  asm volatile("s_wait_asynccnt 0x0" ::: "memory");
}
#endif

__device__ __forceinline__ u16 f2bf(float f) {
  unsigned int u = __float_as_uint(f);
  unsigned int r = (u + 0x7FFFu + ((u >> 16) & 1u)) >> 16;  // RNE
  return (u16)r;
}

__device__ __forceinline__ bf16v make_afrag(bf8v_a lo, bf8v_a hi) {
  return __builtin_shufflevector(lo, hi, 0, 1, 2, 3, 4, 5, 6, 7,
                                 8, 9, 10, 11, 12, 13, 14, 15);
}

// Problem constants
#define NB   4
#define NWN  512
#define NS   64
#define ND   384
#define NH   12
#define NDH  32
#define NM   (NB * NWN * NS)   // 131072 rows
#define NQKV (3 * ND)          // 1152
#define NWIN (NB * NWN)        // 2048 windows

// ---------------------------------------------------------------------------
// Kernel 0: convert + transpose weights to bf16, N-major (K contiguous).
// ---------------------------------------------------------------------------
__global__ __launch_bounds__(256) void cvt_weights(const float* __restrict__ w_qkv,
                                                   const float* __restrict__ w_o,
                                                   u16* __restrict__ wq_t,
                                                   u16* __restrict__ wo_t) {
  const int stride = gridDim.x * 256;
  for (int v = blockIdx.x * 256 + threadIdx.x; v < NQKV * ND; v += stride) {
    const int n = v / ND, k = v % ND;
    wq_t[v] = f2bf(w_qkv[(size_t)k * NQKV + n]);
  }
  for (int v = blockIdx.x * 256 + threadIdx.x; v < ND * ND; v += stride) {
    const int n = v / ND, k = v % ND;
    wo_t[v] = f2bf(w_o[(size_t)k * ND + n]);
  }
}

// ---------------------------------------------------------------------------
// Kernel 1: QKV projection. GEMM M=131072, N=1152, K=384.
// Block tile 128x64, 8 waves, each wave owns a 16x64 strip (4 WMMA n-tiles).
// Weight tile staged via async global->LDS; x tile needs f32->bf16 convert so
// it goes through VGPRs. Epilogue: +bias, scatter Q,K [h][s][dh], V [h][dh][s].
// ---------------------------------------------------------------------------
__global__ __launch_bounds__(256) void qkv_gemm(const float* __restrict__ x,
                                                const u16* __restrict__ wq_t,
                                                const float* __restrict__ b_qkv,
                                                u16* __restrict__ qws,
                                                u16* __restrict__ kws,
                                                u16* __restrict__ vws) {
  __shared__ alignas(64) u16 xs[128][36];  // A tile: 128 rows x 32 K (bf16), padded
  __shared__ alignas(64) u16 wt[64][32];   // B tile: 64 cols x 32 K (bf16)

  const int tid = threadIdx.x;
  const int lane = tid & 31, wave = tid >> 5;
  const int lane15 = lane & 15;
  const int hi8 = (lane >> 4) << 3;
  const int m0 = blockIdx.x * 128;
  const int n0 = blockIdx.y * 64;

  v8f acc[4] = {};

  for (int k0 = 0; k0 < ND; k0 += 32) {
    __syncthreads();
    // stage weight tile (bf16, N-major): 64x32 u16 = 512 x b64, 2 per thread
#pragma unroll
    for (int j = 0; j < 2; ++j) {
      const int v = tid + 256 * j;
      const int n = v >> 3, c = (v & 7) << 2;
      const u16* src = wq_t + (size_t)(n0 + n) * ND + k0 + c;
#if HAS_ASYNC_LDS
      async_copy_b64(src, &wt[n][c]);
#else
      *(uint2*)&wt[n][c] = *(const uint2*)src;
#endif
    }
    // stage x tile (fp32 -> bf16): 128x32 = 1024 float4, 4 per thread
#pragma unroll
    for (int j = 0; j < 4; ++j) {
      const int v = tid + 256 * j;
      const int r = v >> 3, c = (v & 7) << 2;
      const float4 f = *(const float4*)(x + (size_t)(m0 + r) * ND + k0 + c);
      xs[r][c + 0] = f2bf(f.x);
      xs[r][c + 1] = f2bf(f.y);
      xs[r][c + 2] = f2bf(f.z);
      xs[r][c + 3] = f2bf(f.w);
    }
    if (k0 + 32 < ND)  // pull next x tile toward L2/L0 while we do math
      __builtin_prefetch(x + (size_t)(m0 + (tid >> 1)) * ND + k0 + 32, 0, 1);
#if HAS_ASYNC_LDS
    async_wait();
#endif
    __syncthreads();

    // A fragment: row = wave*16 + lane15, chunks at K=16*(lane>>4) and K=16*(lane>>4)+8
    const int mr = wave * 16 + lane15;
    bf8v_a a0 = *(const bf8v_a*)&xs[mr][(lane >> 4) << 4];
    bf8v_a a1 = *(const bf8v_a*)&xs[mr][((lane >> 4) << 4) + 8];
    const bf16v afrag = make_afrag(a0, a1);

#pragma unroll
    for (int nt = 0; nt < 4; ++nt) {
      const bf16v bfrag = *(const bf16v_a*)&wt[nt * 16 + lane15][(lane >> 4) << 4];
      acc[nt] = WMMA_BF16(afrag, bfrag, acc[nt]);
    }
  }

  // epilogue: bias + scatter into per-head Q/K (row-major) and V (transposed)
#pragma unroll
  for (int nt = 0; nt < 4; ++nt) {
    const int gn = n0 + nt * 16 + lane15;  // 0..1151
    const int sec = gn / ND;               // 0=Q 1=K 2=V (uniform per strip)
    const int rn = gn % ND;
    const int hh = rn >> 5, dh = rn & 31;
    const float bias = b_qkv[gn];
#pragma unroll
    for (int i = 0; i < 8; ++i) {
      const int gm = m0 + wave * 16 + hi8 + i;
      const int bw = gm >> 6, s = gm & 63;
      const size_t bh = (size_t)bw * NH + hh;
      const u16 val = f2bf(acc[nt][i] + bias);
      if (sec == 0)
        qws[(bh * NS + s) * NDH + dh] = val;
      else if (sec == 1)
        kws[(bh * NS + s) * NDH + dh] = val;
      else
        vws[(bh * NDH + dh) * NS + s] = val;
    }
  }
}

// ---------------------------------------------------------------------------
// Kernel 2: windowed attention. One block per window, 4 waves x 3 heads each.
// QK^T: 16 WMMAs (K=32 == Dh, single step). Softmax+rel-bias in LDS (P
// overwrites score rows in place as bf16). PV: 16 WMMAs (K=64 = 2 steps).
// ---------------------------------------------------------------------------
__global__ __launch_bounds__(128) void win_attn(const u16* __restrict__ qws,
                                                const u16* __restrict__ kws,
                                                const u16* __restrict__ vws,
                                                const float* __restrict__ pos_bias,
                                                u16* __restrict__ ao) {
  __shared__ alignas(64) float sc[4][64][66];  // padded rows; P aliases row head

  const int tid = threadIdx.x;
  const int lane = tid & 31, wave = tid >> 5;
  const int lane15 = lane & 15;
  const int hi8 = (lane >> 4) << 3;
  const int bw = blockIdx.x;
  const float scale = 0.17677669529663687f;  // 1/sqrt(32)

  for (int it = 0; it < 3; ++it) {
    const int h = wave * 3 + it;
    const size_t bh = (size_t)bw * NH + h;
    const u16* qh = qws + bh * NS * NDH;
    const u16* kh = kws + bh * NS * NDH;
    const u16* vh = vws + bh * NDH * NS;

    // ---- scores = Q K^T -------------------------------------------------
#pragma unroll
    for (int mi = 0; mi < 4; ++mi) {
      const u16* qp = qh + (mi * 16 + lane15) * NDH + ((lane >> 4) << 4);
      bf8v_a a0 = *(const bf8v_a*)qp;
      bf8v_a a1 = *(const bf8v_a*)(qp + 8);
      const bf16v afrag = make_afrag(a0, a1);
#pragma unroll
      for (int ni = 0; ni < 4; ++ni) {
        const bf16v bfrag =
            *(const bf16v_a*)(kh + (ni * 16 + lane15) * NDH + ((lane >> 4) << 4));
        v8f s = {};
        s = WMMA_BF16(afrag, bfrag, s);
        const int mb = mi * 16 + hi8;
        const int nc = ni * 16 + lane15;
#pragma unroll
        for (int i = 0; i < 8; ++i) sc[wave][mb + i][nc] = s[i];
      }
    }
    __syncthreads();

    // ---- softmax with relative-position bias (rows lane, lane+32) -------
#pragma unroll
    for (int rr = 0; rr < 2; ++rr) {
      const int r = lane + rr * 32;
      float* row = &sc[wave][r][0];
      float mx = -3.0e30f;
      for (int c = 0; c < 64; ++c) {
        const float t = row[c] * scale + pos_bias[(c - r + 63) * NH + h];
        mx = fmaxf(mx, t);
      }
      float sum = 0.0f;
      for (int c = 0; c < 64; ++c) {
        const float t = row[c] * scale + pos_bias[(c - r + 63) * NH + h];
        sum += __expf(t - mx);
      }
      const float inv = 1.0f / sum;
      u16_a* prow = (u16_a*)row;  // bf16 P overlays the f32 row (write < read)
      for (int c = 0; c < 64; ++c) {
        const float t = row[c] * scale + pos_bias[(c - r + 63) * NH + h];
        prow[c] = f2bf(__expf(t - mx) * inv);
      }
    }
    __syncthreads();

    // ---- out = P V ------------------------------------------------------
#pragma unroll
    for (int mi = 0; mi < 4; ++mi) {
      const u16_a* prow = (const u16_a*)&sc[wave][mi * 16 + lane15][0];
      bf16v afrag[2];
#pragma unroll
      for (int kk = 0; kk < 2; ++kk) {
        const int kb = kk * 32 + ((lane >> 4) << 4);
        bf8v_a a0 = *(const bf8v_a*)(prow + kb);
        bf8v_a a1 = *(const bf8v_a*)(prow + kb + 8);
        afrag[kk] = make_afrag(a0, a1);
      }
#pragma unroll
      for (int ni = 0; ni < 2; ++ni) {
        v8f o = {};
#pragma unroll
        for (int kk = 0; kk < 2; ++kk) {
          const bf16v bfrag = *(const bf16v_a*)(vh + (ni * 16 + lane15) * NS +
                                                kk * 32 + ((lane >> 4) << 4));
          o = WMMA_BF16(afrag[kk], bfrag, o);
        }
        const int mb = mi * 16 + hi8;
        const int nc = ni * 16 + lane15;
#pragma unroll
        for (int i = 0; i < 8; ++i)
          ao[((size_t)bw * NS + mb + i) * ND + h * NDH + nc] = f2bf(o[i]);
      }
    }
    __syncthreads();
  }
}

// ---------------------------------------------------------------------------
// Kernel 3: output projection. GEMM M=131072, N=384, K=384, f32 out + bias.
// Both tiles are already bf16 -> fully async global->LDS staging.
// ---------------------------------------------------------------------------
__global__ __launch_bounds__(256) void out_proj(const u16* __restrict__ ao,
                                                const u16* __restrict__ wo_t,
                                                const float* __restrict__ b_o,
                                                float* __restrict__ out) {
  __shared__ alignas(64) u16 as_[128][32];
  __shared__ alignas(64) u16 bs_[64][32];

  const int tid = threadIdx.x;
  const int lane = tid & 31, wave = tid >> 5;
  const int lane15 = lane & 15;
  const int hi8 = (lane >> 4) << 3;
  const int m0 = blockIdx.x * 128;
  const int n0 = blockIdx.y * 64;

  v8f acc[4] = {};

  for (int k0 = 0; k0 < ND; k0 += 32) {
    __syncthreads();
    // A tile: 128x32 u16 = 1024 x b64, 4 per thread
#pragma unroll
    for (int j = 0; j < 4; ++j) {
      const int v = tid + 256 * j;
      const int r = v >> 3, c = (v & 7) << 2;
      const u16* src = ao + (size_t)(m0 + r) * ND + k0 + c;
#if HAS_ASYNC_LDS
      async_copy_b64(src, &as_[r][c]);
#else
      *(uint2*)&as_[r][c] = *(const uint2*)src;
#endif
    }
    // B tile: 64x32 u16 = 512 x b64, 2 per thread
#pragma unroll
    for (int j = 0; j < 2; ++j) {
      const int v = tid + 256 * j;
      const int n = v >> 3, c = (v & 7) << 2;
      const u16* src = wo_t + (size_t)(n0 + n) * ND + k0 + c;
#if HAS_ASYNC_LDS
      async_copy_b64(src, &bs_[n][c]);
#else
      *(uint2*)&bs_[n][c] = *(const uint2*)src;
#endif
    }
    if (k0 + 32 < ND)
      __builtin_prefetch(ao + (size_t)(m0 + (tid >> 1)) * ND + k0 + 32, 0, 1);
#if HAS_ASYNC_LDS
    async_wait();
#endif
    __syncthreads();

    const int mr = wave * 16 + lane15;
    bf8v_a a0 = *(const bf8v_a*)&as_[mr][(lane >> 4) << 4];
    bf8v_a a1 = *(const bf8v_a*)&as_[mr][((lane >> 4) << 4) + 8];
    const bf16v afrag = make_afrag(a0, a1);
#pragma unroll
    for (int nt = 0; nt < 4; ++nt) {
      const bf16v bfrag = *(const bf16v_a*)&bs_[nt * 16 + lane15][(lane >> 4) << 4];
      acc[nt] = WMMA_BF16(afrag, bfrag, acc[nt]);
    }
  }

#pragma unroll
  for (int nt = 0; nt < 4; ++nt) {
    const int gn = n0 + nt * 16 + lane15;
    const float bias = b_o[gn];
#pragma unroll
    for (int i = 0; i < 8; ++i) {
      const int gm = m0 + wave * 16 + hi8 + i;
      out[(size_t)gm * ND + gn] = acc[nt][i] + bias;
    }
  }
}

// ---------------------------------------------------------------------------
extern "C" void kernel_launch(void* const* d_in, const int* in_sizes, int n_in,
                              void* d_out, int out_size, void* d_ws, size_t ws_size,
                              hipStream_t stream) {
  const float* x        = (const float*)d_in[0];
  const float* w_qkv    = (const float*)d_in[1];
  const float* b_qkv    = (const float*)d_in[2];
  const float* pos_bias = (const float*)d_in[3];
  const float* w_o      = (const float*)d_in[4];
  const float* b_o      = (const float*)d_in[5];
  float* out = (float*)d_out;

  char* ws = (char*)d_ws;
  size_t off = 0;
  auto carve = [&](size_t bytes) -> void* {
    off = (off + 255) & ~(size_t)255;
    void* p = ws + off;
    off += bytes;
    return p;
  };

  u16* wq_t = (u16*)carve((size_t)NQKV * ND * sizeof(u16));
  u16* wo_t = (u16*)carve((size_t)ND * ND * sizeof(u16));
  u16* qws  = (u16*)carve((size_t)NWIN * NH * NS * NDH * sizeof(u16));
  u16* kws  = (u16*)carve((size_t)NWIN * NH * NS * NDH * sizeof(u16));
  u16* vws  = (u16*)carve((size_t)NWIN * NH * NS * NDH * sizeof(u16));
  u16* aows = (u16*)carve((size_t)NM * ND * sizeof(u16));

  cvt_weights<<<dim3(512), dim3(256), 0, stream>>>(w_qkv, w_o, wq_t, wo_t);
  qkv_gemm<<<dim3(NM / 128, NQKV / 64), dim3(256), 0, stream>>>(x, wq_t, b_qkv,
                                                                qws, kws, vws);
  win_attn<<<dim3(NWIN), dim3(128), 0, stream>>>(qws, kws, vws, pos_bias, aows);
  out_proj<<<dim3(NM / 128, ND / 64), dim3(256), 0, stream>>>(aows, wo_t, b_o, out);
}
